// GCN_44306882625938
// MI455X (gfx1250) — matrix operand order repaired and checked by reading, not compiled, not run on hardware
//
#include <hip/hip_runtime.h>

typedef __attribute__((ext_vector_type(16))) _Float16 v16h;
typedef __attribute__((ext_vector_type(8)))  _Float16 v8h;
typedef __attribute__((ext_vector_type(4)))  _Float16 v4h;
typedef __attribute__((ext_vector_type(8)))  float    v8f;

#define GCN_D 512

// ---------------------------------------------------------------------------
// Native CDNA5 no-return fp32 global atomic add (STOREcnt-tracked, DEV scope).
// ---------------------------------------------------------------------------
__device__ __forceinline__ void atomic_add_f32_dev(float* p, float v) {
    asm volatile("global_atomic_add_f32 %0, %1, off scope:SCOPE_DEV"
                 :: "v"(p), "v"(v)
                 : "memory");
}

// ---------------------------------------------------------------------------
// CDNA5 async global -> LDS copy (ASYNCcnt-tracked) + wait.
// lds_addr is the 32-bit LDS byte offset (low 32 bits of the flat address).
// ---------------------------------------------------------------------------
__device__ __forceinline__ void async_g2l_b128(unsigned lds_addr, const void* gaddr) {
    asm volatile("global_load_async_to_lds_b128 %0, %1, off"
                 :: "v"(lds_addr), "v"(gaddr)
                 : "memory");
}
__device__ __forceinline__ void wait_asynccnt0() {
    asm volatile("s_wait_asynccnt 0x0" ::: "memory");
}

// ---------------------------------------------------------------------------
// Kernel 1: W [D][D] f32 row-major  ->  Wt [D][D] f16 transposed (Wt[n][k]).
// ---------------------------------------------------------------------------
__global__ void gcn_wt_convert(const float* __restrict__ W,
                               _Float16* __restrict__ Wt) {
    int i = blockIdx.x * blockDim.x + threadIdx.x;   // i = k*512 + n
    int k = i >> 9;
    int n = i & 511;
    Wt[((size_t)n << 9) + k] = (_Float16)W[i];
}

// ---------------------------------------------------------------------------
// Kernel 2 (fused): out = x + b  AND  xh = f16(x)   (one pass over x)
// ---------------------------------------------------------------------------
__global__ void gcn_init(const float* __restrict__ x,
                         const float* __restrict__ b,
                         float* __restrict__ out,
                         _Float16* __restrict__ xh, long n4) {
    long i = (long)blockIdx.x * blockDim.x + threadIdx.x;
    if (i >= n4) return;
    float4 xv = ((const float4*)x)[i];
    float4 bv = ((const float4*)b)[i & 127];
    float4 o;
    o.x = xv.x + bv.x; o.y = xv.y + bv.y;
    o.z = xv.z + bv.z; o.w = xv.w + bv.w;
    ((float4*)out)[i] = o;
    v4h hx;
    hx[0] = (_Float16)xv.x; hx[1] = (_Float16)xv.y;
    hx[2] = (_Float16)xv.z; hx[3] = (_Float16)xv.w;
    ((v4h*)xh)[i] = hx;
}

// ---------------------------------------------------------------------------
// Kernel 3: support(f16) = xh(f16) @ W  via v_wmma_f32_16x16x32_f16.
// Block: 256 thr = 8 waves = 8 row-tiles x one 64-col group.
// Wt slab (64 cols x 512 K = 64 KB, contiguous in Wt) async-copied to LDS.
// ---------------------------------------------------------------------------
__global__ __launch_bounds__(256)
void gcn_gemm_wmma(const _Float16* __restrict__ xh,
                   const _Float16* __restrict__ Wt,
                   _Float16* __restrict__ support, int N) {
    const int D = GCN_D;
    __shared__ _Float16 sW[64 * GCN_D];               // 64 KB

    int colGrp = blockIdx.x & 7;                      // 8 groups of 64 cols
    int rowGrp = blockIdx.x >> 3;                     // 8 row-tiles per block
    int col0 = colGrp << 6;

    // ---- async copy Wt[col0 .. col0+63][0..511] (contiguous 64 KB) -> LDS
    {
        const char* gsrc = (const char*)(Wt + ((size_t)col0 << 9));
        unsigned lbase = (unsigned)(uintptr_t)(&sW[0]) + threadIdx.x * 16u;
        const char* g = gsrc + threadIdx.x * 16;
#pragma unroll
        for (int r = 0; r < 16; ++r)
            async_g2l_b128(lbase + (unsigned)(r * 4096), g + r * 4096);
    }
    wait_asynccnt0();
    __syncthreads();

    int wave = threadIdx.x >> 5;
    int lane = threadIdx.x & 31;
    int rowTile = (rowGrp << 3) + wave;
    int row0 = rowTile << 4;
    if (row0 < N) {
        int m = lane & 15;
        int h = lane >> 4;

        int arow = row0 + m;
        if (arow >= N) arow = N - 1;                  // tail clamp
        const _Float16* xrow = xh + (size_t)arow * D;
        // B fragment base for this lane: column (col0 + t*16 + m), K pattern 16h+j
        const _Float16* sb = &sW[((unsigned)m << 9) + (unsigned)(16 * h)];

        v8f acc0 = {}, acc1 = {}, acc2 = {}, acc3 = {};

        for (int k0 = 0; k0 < D; k0 += 32) {
            // A frag: element j -> K=k0+16h+j
            v8h a0 = *(const v8h*)(xrow + k0 + 16 * h);
            v8h a1 = *(const v8h*)(xrow + k0 + 16 * h + 8);
            v16h A;
#pragma unroll
            for (int j = 0; j < 8; ++j) { A[j] = a0[j]; A[8 + j] = a1[j]; }

            v16h B0, B1, B2, B3;
#pragma unroll
            for (int j = 0; j < 8; ++j) {
                v8h t0 = *(const v8h*)(sb + (0 << 13) + k0);      // (t*16)<<9
                v8h t1 = *(const v8h*)(sb + (0 << 13) + k0 + 8);
                B0[j] = t0[j]; B0[8 + j] = t1[j];
            }
            {
                const _Float16* p = sb + (16 << 9) + k0;
                v8h t0 = *(const v8h*)p, t1 = *(const v8h*)(p + 8);
#pragma unroll
                for (int j = 0; j < 8; ++j) { B1[j] = t0[j]; B1[8 + j] = t1[j]; }
            }
            {
                const _Float16* p = sb + (32 << 9) + k0;
                v8h t0 = *(const v8h*)p, t1 = *(const v8h*)(p + 8);
#pragma unroll
                for (int j = 0; j < 8; ++j) { B2[j] = t0[j]; B2[8 + j] = t1[j]; }
            }
            {
                const _Float16* p = sb + (48 << 9) + k0;
                v8h t0 = *(const v8h*)p, t1 = *(const v8h*)(p + 8);
#pragma unroll
                for (int j = 0; j < 8; ++j) { B3[j] = t0[j]; B3[8 + j] = t1[j]; }
            }

            acc0 = __builtin_amdgcn_wmma_f32_16x16x32_f16(false, A, false, B0,
                                                          (short)0, acc0, false, false);
            acc1 = __builtin_amdgcn_wmma_f32_16x16x32_f16(false, A, false, B1,
                                                          (short)0, acc1, false, false);
            acc2 = __builtin_amdgcn_wmma_f32_16x16x32_f16(false, A, false, B2,
                                                          (short)0, acc2, false, false);
            acc3 = __builtin_amdgcn_wmma_f32_16x16x32_f16(false, A, false, B3,
                                                          (short)0, acc3, false, false);
        }

        // D layout: VGPR g -> row row0+8h+g, lane -> col col0 + t*16 + m
#pragma unroll
        for (int g = 0; g < 8; ++g) {
            int r = row0 + 8 * h + g;
            if (r < N) {
                size_t base = (size_t)r * D + col0 + m;
                support[base +  0] = (_Float16)acc0[g];
                support[base + 16] = (_Float16)acc1[g];
                support[base + 32] = (_Float16)acc2[g];
                support[base + 48] = (_Float16)acc3[g];
            }
        }
    }
}

// ---------------------------------------------------------------------------
// Kernel 4: per-edge gather (f16 support row), scale, scatter-add (f32 atomics)
// One wave per edge; each lane handles 16 consecutive features (32 B).
// ---------------------------------------------------------------------------
__global__ __launch_bounds__(256)
void gcn_edge_scatter(const int* __restrict__ esrc,
                      const int* __restrict__ edst,
                      const float* __restrict__ ew,
                      const _Float16* __restrict__ support,
                      float* __restrict__ out, int E) {
    int wave = (int)((blockIdx.x * blockDim.x + threadIdx.x) >> 5);
    if (wave >= E) return;
    int lane = threadIdx.x & 31;
    int s = esrc[wave];
    int d = edst[wave];
    float w = ew[wave];

    const _Float16* sp = support + ((size_t)s << 9) + lane * 16;
    float*          op = out     + ((size_t)d << 9) + lane * 16;

    v8h h0 = *(const v8h*)(sp);
    v8h h1 = *(const v8h*)(sp + 8);
#pragma unroll
    for (int j = 0; j < 8; ++j)
        atomic_add_f32_dev(op + j, (float)h0[j] * w);
#pragma unroll
    for (int j = 0; j < 8; ++j)
        atomic_add_f32_dev(op + 8 + j, (float)h1[j] * w);
}

// ---------------------------------------------------------------------------
// Kernel 5: out = tanh(out)  (float4 vectorized, in place)
// ---------------------------------------------------------------------------
__global__ void gcn_tanh(float* __restrict__ out, long n4) {
    long i = (long)blockIdx.x * blockDim.x + threadIdx.x;
    if (i >= n4) return;
    float4 v = ((float4*)out)[i];
    v.x = tanhf(v.x); v.y = tanhf(v.y);
    v.z = tanhf(v.z); v.w = tanhf(v.w);
    ((float4*)out)[i] = v;
}

// ---------------------------------------------------------------------------
extern "C" void kernel_launch(void* const* d_in, const int* in_sizes, int n_in,
                              void* d_out, int out_size, void* d_ws, size_t ws_size,
                              hipStream_t stream) {
    const int D = GCN_D;
    const float* x    = (const float*)d_in[0];
    const int*   esrc = (const int*)  d_in[1];
    const int*   edst = (const int*)  d_in[2];
    const float* ew   = (const float*)d_in[3];
    const float* W    = (const float*)d_in[4];
    const float* b    = (const float*)d_in[5];
    float* out = (float*)d_out;

    int N = in_sizes[0] / D;      // 100000
    int E = in_sizes[1];          // 3200000

    // Workspace layout: [ support f16: N*D ][ xh f16: N*D ][ Wt f16: D*D ]
    size_t ndBytes = ((size_t)N * D * sizeof(_Float16) + 255) & ~(size_t)255;
    _Float16* support = (_Float16*)d_ws;
    _Float16* xh      = (_Float16*)((char*)d_ws + ndBytes);
    _Float16* Wt      = (_Float16*)((char*)d_ws + 2 * ndBytes);

    // 1) W -> Wt (f16, transposed)
    gcn_wt_convert<<<(D * D) / 256, 256, 0, stream>>>(W, Wt);

    // 2) out = x + b ; xh = f16(x)
    long n4 = (long)N * D / 4;
    {
        long blocks = (n4 + 255) / 256;
        gcn_init<<<(unsigned)blocks, 256, 0, stream>>>(x, b, out, xh, n4);
    }
    // 3) support = f16(xh @ W)  (WMMA + async LDS staging)
    {
        int rowGrps = (N + 127) / 128;       // 8 row-tiles (128 rows) per block
        unsigned blocks = (unsigned)rowGrps * 8u;   // x 8 col-groups
        gcn_gemm_wmma<<<blocks, 256, 0, stream>>>(xh, Wt, support, N);
    }
    // 4) out += scatter(support[src] * w -> dst)
    {
        long threads = (long)E * 32;
        long blocks  = (threads + 255) / 256;
        gcn_edge_scatter<<<(unsigned)blocks, 256, 0, stream>>>(esrc, edst, ew,
                                                               support, out, E);
    }
    // 5) out = tanh(out)
    {
        long blocks = (n4 + 255) / 256;
        gcn_tanh<<<(unsigned)blocks, 256, 0, stream>>>(out, n4);
    }
}
